// KimiDecoderLayer_68195490726079
// MI455X (gfx1250) — matrix operand order repaired and checked by reading, not compiled, not run on hardware
//
#include <hip/hip_runtime.h>

// ---------------- problem constants ----------------
constexpr int Bc = 2, Tc = 1024, Hc = 2048, NHc = 8, Dc = 128, Pc = 1024, Kc = 4;
constexpr int NTOK = Bc * Tc;            // 2048
constexpr float EPSc = 1e-6f;
constexpr float SCALEc = 0.0883883476483184406f; // 128^-0.5

typedef _Float16 f16_t;
typedef __attribute__((ext_vector_type(16))) _Float16 v16h;
typedef __attribute__((ext_vector_type(8)))  _Float16 v8h;
typedef __attribute__((ext_vector_type(8)))  float    v8f;
typedef __attribute__((ext_vector_type(4)))  int      v4i;

// address-space qualified pointer types for async global->LDS copies
typedef __attribute__((address_space(1))) v4i* gv4i_p;   // global
typedef __attribute__((address_space(3))) v4i* lv4i_p;   // LDS

union Afrag { v16h v; v8h h[2]; };

#if defined(__has_builtin)
#if __has_builtin(__builtin_amdgcn_global_load_async_to_lds_b128) && \
    __has_builtin(__builtin_amdgcn_s_wait_asynccnt)
#define HAVE_ASYNC_LDS 1
#endif
#endif
#ifndef HAVE_ASYNC_LDS
#define HAVE_ASYNC_LDS 0
#endif

// ---------------- f32 -> f16 convert ----------------
__global__ void cvt_f32_f16(const float* __restrict__ in, f16_t* __restrict__ out, long n) {
    long i = (long)blockIdx.x * blockDim.x + threadIdx.x;
    if (i < n) out[i] = (f16_t)in[i];
}

// ---------------- WMMA GEMM: C[M,N] = A[M,K] * Bw[N,K]^T ----------------
// A,Bw f16 row-major (K contiguous). Writes f32 to Cf or f16 to Ch (one non-null).
// Block 256 thr = 8 waves; wave (wm 0..3, wn 0..1) owns 32x64; block tile 128x128x32.
// A/B tiles double-buffered in LDS via async global->LDS copies.
__global__ __launch_bounds__(256)
void gemm_wmma_f16(const f16_t* __restrict__ A, const f16_t* __restrict__ Bw,
                   float* __restrict__ Cf, f16_t* __restrict__ Ch,
                   int M, int N, int Kd) {
    __shared__ f16_t As[2][128 * 32];
    __shared__ f16_t Bs[2][128 * 32];

    const int tid  = threadIdx.x;
    const int lane = tid & 31;
    const int wave = tid >> 5;
    const int l16  = lane & 15;
    const int hi   = lane >> 4;          // 0/1 half-wave
    const int wm   = wave >> 1;          // 0..3
    const int wn   = wave & 1;           // 0..1
    const int mblk = blockIdx.y * 128;
    const int nblk = blockIdx.x * 128;

    v8f acc[2][4];
    #pragma unroll
    for (int i = 0; i < 2; ++i)
        #pragma unroll
        for (int j = 0; j < 4; ++j)
            #pragma unroll
            for (int e = 0; e < 8; ++e) acc[i][j][e] = 0.0f;

    // each thread moves 2x16B of A and 2x16B of B per K-step (8KB + 8KB tiles)
    auto issue_copy = [&](int p, int kk) {
        #pragma unroll
        for (int r = 0; r < 2; ++r) {
            int chunk = tid + r * 256;        // 0..511
            int row   = chunk >> 2;           // 0..127
            int cpos  = (chunk & 3) * 8;      // 0,8,16,24 (f16 elems)
            const f16_t* ga = A  + (long)(mblk + row) * Kd + kk + cpos;
            const f16_t* gb = Bw + (long)(nblk + row) * Kd + kk + cpos;
            f16_t* la = &As[p][row * 32 + cpos];
            f16_t* lb = &Bs[p][row * 32 + cpos];
#if HAVE_ASYNC_LDS
            __builtin_amdgcn_global_load_async_to_lds_b128((gv4i_p)ga, (lv4i_p)la, 0, 0);
            __builtin_amdgcn_global_load_async_to_lds_b128((gv4i_p)gb, (lv4i_p)lb, 0, 0);
#else
            *(v8h*)la = *(const v8h*)ga;
            *(v8h*)lb = *(const v8h*)gb;
#endif
        }
    };

    issue_copy(0, 0);
    const int nk = Kd >> 5;             // K-steps of 32
    for (int kt = 0; kt < nk; ++kt) {
        const int p = kt & 1;
#if HAVE_ASYNC_LDS
        __builtin_amdgcn_s_wait_asynccnt(0);   // own wave's copies into buf p done
#endif
        __syncthreads();                       // all waves' copies done; buf 1-p free
        if (kt + 1 < nk) issue_copy(1 - p, (kt + 1) * 32);

        Afrag a[2];
        #pragma unroll
        for (int mt = 0; mt < 2; ++mt) {
            const f16_t* rp = &As[p][(wm * 32 + mt * 16 + l16) * 32 + hi * 8];
            a[mt].h[0] = *(const v8h*)(rp);        // K[hi*8 .. +7]
            a[mt].h[1] = *(const v8h*)(rp + 16);   // K[16+hi*8 .. +7]
        }
        v16h bf[4];
        #pragma unroll
        for (int nt = 0; nt < 4; ++nt) {
            const f16_t* rp = &Bs[p][(wn * 64 + nt * 16 + l16) * 32 + hi * 8];
            Afrag t; t.h[0] = *(const v8h*)(rp); t.h[1] = *(const v8h*)(rp + 16);
            bf[nt] = t.v;                          // K[hi*8 .. +7], K[16+hi*8 .. +7]
        }
        #pragma unroll
        for (int mt = 0; mt < 2; ++mt)
            #pragma unroll
            for (int nt = 0; nt < 4; ++nt)
                acc[mt][nt] = __builtin_amdgcn_wmma_f32_16x16x32_f16(
                    false, a[mt].v, false, bf[nt], (short)0, acc[mt][nt], false, false);
    }

    // C/D layout: vgpr j: lanes0-15 row=j, lanes16-31 row=j+8; col=l16
    const int mBase = mblk + wm * 32 + hi * 8;
    const int nBase = nblk + wn * 64 + l16;
    if (Cf) {
        #pragma unroll
        for (int mt = 0; mt < 2; ++mt)
            #pragma unroll
            for (int nt = 0; nt < 4; ++nt)
                #pragma unroll
                for (int j = 0; j < 8; ++j)
                    Cf[(long)(mBase + mt * 16 + j) * N + nBase + nt * 16] = acc[mt][nt][j];
    } else {
        #pragma unroll
        for (int mt = 0; mt < 2; ++mt)
            #pragma unroll
            for (int nt = 0; nt < 4; ++nt)
                #pragma unroll
                for (int j = 0; j < 8; ++j)
                    Ch[(long)(mBase + mt * 16 + j) * N + nBase + nt * 16] =
                        (f16_t)acc[mt][nt][j];
    }
}

// ---------------- causal depthwise conv (K=4) + SiLU (+optional L2 norm) ----------------
// raw: (B,T,P) f32 ; w: (P,K) ; out: (B,T,P). mode: 0=v, 1=k (l2norm), 2=q (l2norm*scale)
__global__ __launch_bounds__(128)
void conv_silu_norm(const float* __restrict__ raw, const float* __restrict__ w,
                    float* __restrict__ out, int mode) {
    const int bth = blockIdx.x;           // (b*T + t)*NH + h
    const int h  = bth % NHc;
    const int bt = bth / NHc;             // b*T + t
    const int tt = bt % Tc;
    const int d  = threadIdx.x;           // 0..127
    const int c  = h * Dc + d;

    float s = 0.0f;
    #pragma unroll
    for (int j = 0; j < Kc; ++j) {
        int tsrc = tt - (Kc - 1) + j;
        float xv = (tsrc >= 0) ? raw[((long)(bt - tt) + tsrc) * Pc + c] : 0.0f;
        s += w[c * Kc + j] * xv;
    }
    float y = s * (1.0f / (1.0f + __expf(-s)));   // silu

    if (mode) {
        float sq = y * y;
        #pragma unroll
        for (int off = 16; off; off >>= 1) sq += __shfl_xor(sq, off, 32);
        __shared__ float red[4];
        if ((threadIdx.x & 31) == 0) red[threadIdx.x >> 5] = sq;
        __syncthreads();
        float tot = red[0] + red[1] + red[2] + red[3];
        y *= rsqrtf(tot + EPSc);
        if (mode == 2) y *= SCALEc;
    }
    out[(long)bt * Pc + c] = y;
}

// ---------------- gate decay: g = -exp(A_log[h]) * softplus(gf + dt_bias) ----------------
__global__ void gate_decay(const float* __restrict__ gf, const float* __restrict__ A_log,
                           const float* __restrict__ dt_bias, float* __restrict__ g, long n) {
    long i = (long)blockIdx.x * blockDim.x + threadIdx.x;
    if (i >= n) return;
    int c = (int)(i % Pc);
    int h = c / Dc;
    float x = gf[i] + dt_bias[c];
    float sp = (x > 20.0f) ? x : log1pf(__expf(x));
    g[i] = -__expf(A_log[h]) * sp;
}

// ---------------- beta = sigmoid(x @ Wb^T), N=8 (tiny) ----------------
__global__ __launch_bounds__(256)
void beta_kernel(const float* __restrict__ x, const float* __restrict__ Wb,
                 float* __restrict__ beta) {
    int gw   = blockIdx.x * 8 + (threadIdx.x >> 5);   // one wave per (m,h)
    int lane = threadIdx.x & 31;
    int m = gw >> 3, h = gw & 7;
    const float* xr = x  + (long)m * Hc;
    const float* wr = Wb + (long)h * Hc;
    float s = 0.0f;
    for (int j = lane; j < Hc; j += 32) s += xr[j] * wr[j];
    #pragma unroll
    for (int off = 16; off; off >>= 1) s += __shfl_xor(s, off, 32);
    if (lane == 0) beta[gw] = 1.0f / (1.0f + __expf(-s));
}

// ---------------- gated delta-rule recurrence ----------------
// One block per (b,h); thread t owns value column dv=t; S[dk] kept in VGPRs.
__global__ __launch_bounds__(128)
void kda_recurrence(const float* __restrict__ q, const float* __restrict__ k,
                    const float* __restrict__ v, const float* __restrict__ g,
                    const float* __restrict__ beta, float* __restrict__ o) {
    const int bh = blockIdx.x;
    const int b = bh >> 3, h = bh & 7;
    const int t = threadIdx.x;            // dv
    __shared__ float lq[Dc], lk[Dc], leg[Dc];
    __shared__ float lbeta;

    float S[Dc];
    #pragma unroll
    for (int i = 0; i < Dc; ++i) S[i] = 0.0f;

    const long strideT = (long)NHc * Dc;                    // 1024
    const long base = (long)b * Tc * strideT + (long)h * Dc;

    for (int step = 0; step < Tc; ++step) {
        long idx = base + (long)step * strideT + t;
        float qv   = q[idx];
        float kv_  = k[idx];
        float gv   = g[idx];
        float vown = v[idx];
        __syncthreads();                       // previous-step LDS reads complete
        lq[t] = qv; lk[t] = kv_; leg[t] = __expf(gv);
        if (t == 0) lbeta = beta[((long)b * Tc + step) * NHc + h];
        __syncthreads();

        float kvacc = 0.0f;
        #pragma unroll
        for (int dk = 0; dk < Dc; ++dk) {      // decay + k^T S
            S[dk] *= leg[dk];
            kvacc += lk[dk] * S[dk];
        }
        float delta = (vown - kvacc) * lbeta;
        float oacc = 0.0f;
        #pragma unroll
        for (int dk = 0; dk < Dc; ++dk) {      // rank-1 update + q^T S
            S[dk] += lk[dk] * delta;
            oacc  += lq[dk] * S[dk];
        }
        o[idx] = oacc;
    }
}

// ---------------- gated RMSNorm epilogue -> f16 ----------------
__global__ __launch_bounds__(128)
void out_gate_norm(const float* __restrict__ o, const float* __restrict__ gate,
                   const float* __restrict__ nw, f16_t* __restrict__ og) {
    int bth = blockIdx.x;                 // (b*T + t)*NH + h
    int d = threadIdx.x;
    long idx = (long)bth * Dc + d;
    float gt  = gate[idx];
    float val = o[idx] * (1.0f / (1.0f + __expf(-gt)));
    float sq = val * val;
    #pragma unroll
    for (int off = 16; off; off >>= 1) sq += __shfl_xor(sq, off, 32);
    __shared__ float red[4];
    if ((threadIdx.x & 31) == 0) red[threadIdx.x >> 5] = sq;
    __syncthreads();
    float ms = (red[0] + red[1] + red[2] + red[3]) * (1.0f / (float)Dc);
    val = val * rsqrtf(ms + EPSc) * nw[d];
    og[idx] = (f16_t)val;
}

// ---------------- host orchestration ----------------
extern "C" void kernel_launch(void* const* d_in, const int* in_sizes, int n_in,
                              void* d_out, int out_size, void* d_ws, size_t ws_size,
                              hipStream_t stream) {
    const float* x       = (const float*)d_in[0];
    const float* Wq      = (const float*)d_in[1];
    const float* Wk      = (const float*)d_in[2];
    const float* Wv      = (const float*)d_in[3];
    const float* conv_q  = (const float*)d_in[4];
    const float* conv_k  = (const float*)d_in[5];
    const float* conv_v  = (const float*)d_in[6];
    const float* A_log   = (const float*)d_in[7];
    const float* dt_bias = (const float*)d_in[8];
    const float* Wfa     = (const float*)d_in[9];
    const float* Wfb     = (const float*)d_in[10];
    const float* Wb      = (const float*)d_in[11];
    const float* Wga     = (const float*)d_in[12];
    const float* Wgb     = (const float*)d_in[13];
    const float* norm_w  = (const float*)d_in[14];
    const float* Wo      = (const float*)d_in[15];
    float* out = (float*)d_out;

    char* ws = (char*)d_ws;
    size_t off = 0;
    auto alloc = [&](size_t bytes) -> char* {
        char* p = ws + off;
        off += (bytes + 255) & ~(size_t)255;
        return p;
    };

    // f16 staging
    f16_t* xh   = (f16_t*)alloc((size_t)NTOK * Hc * 2);
    f16_t* wqh  = (f16_t*)alloc((size_t)Pc * Hc * 2);
    f16_t* wkh  = (f16_t*)alloc((size_t)Pc * Hc * 2);
    f16_t* wvh  = (f16_t*)alloc((size_t)Pc * Hc * 2);
    f16_t* wfah = (f16_t*)alloc((size_t)Dc * Hc * 2);
    f16_t* wfbh = (f16_t*)alloc((size_t)Pc * Dc * 2);
    f16_t* wgah = (f16_t*)alloc((size_t)Dc * Hc * 2);
    f16_t* wgbh = (f16_t*)alloc((size_t)Pc * Dc * 2);
    f16_t* woh  = (f16_t*)alloc((size_t)Hc * Pc * 2);
    f16_t* xah  = (f16_t*)alloc((size_t)NTOK * Dc * 2);
    f16_t* xgh  = (f16_t*)alloc((size_t)NTOK * Dc * 2);
    f16_t* ogh  = (f16_t*)alloc((size_t)NTOK * Pc * 2);
    // f32 activations
    float* qraw = (float*)alloc((size_t)NTOK * Pc * 4);
    float* kraw = (float*)alloc((size_t)NTOK * Pc * 4);
    float* vraw = (float*)alloc((size_t)NTOK * Pc * 4);
    float* qn   = (float*)alloc((size_t)NTOK * Pc * 4);
    float* kn   = (float*)alloc((size_t)NTOK * Pc * 4);
    float* vn   = (float*)alloc((size_t)NTOK * Pc * 4);
    float* gf   = (float*)alloc((size_t)NTOK * Pc * 4);
    float* gdec = (float*)alloc((size_t)NTOK * Pc * 4);
    float* gate = (float*)alloc((size_t)NTOK * Pc * 4);
    float* beta = (float*)alloc((size_t)NTOK * NHc * 4);
    float* obuf = (float*)alloc((size_t)NTOK * Pc * 4);

    auto cvt = [&](const float* src, f16_t* dst, long n) {
        cvt_f32_f16<<<dim3((unsigned)((n + 255) / 256)), dim3(256), 0, stream>>>(src, dst, n);
    };
    auto gemm = [&](const f16_t* A, const f16_t* Bm, float* Cf, f16_t* Ch,
                    int M, int N, int Kd) {
        dim3 grid((unsigned)(N / 128), (unsigned)(M / 128));
        gemm_wmma_f16<<<grid, dim3(256), 0, stream>>>(A, Bm, Cf, Ch, M, N, Kd);
    };

    // 1) convert activations + weights to f16
    cvt(x,   xh,   (long)NTOK * Hc);
    cvt(Wq,  wqh,  (long)Pc * Hc);
    cvt(Wk,  wkh,  (long)Pc * Hc);
    cvt(Wv,  wvh,  (long)Pc * Hc);
    cvt(Wfa, wfah, (long)Dc * Hc);
    cvt(Wfb, wfbh, (long)Pc * Dc);
    cvt(Wga, wgah, (long)Dc * Hc);
    cvt(Wgb, wgbh, (long)Pc * Dc);
    cvt(Wo,  woh,  (long)Hc * Pc);

    // 2) q/k/v projections (WMMA)
    gemm(xh, wqh, qraw, nullptr, NTOK, Pc, Hc);
    gemm(xh, wkh, kraw, nullptr, NTOK, Pc, Hc);
    gemm(xh, wvh, vraw, nullptr, NTOK, Pc, Hc);

    // 3) causal conv + silu (+ l2norm for q/k)
    conv_silu_norm<<<dim3(NTOK * NHc), dim3(128), 0, stream>>>(qraw, conv_q, qn, 2);
    conv_silu_norm<<<dim3(NTOK * NHc), dim3(128), 0, stream>>>(kraw, conv_k, kn, 1);
    conv_silu_norm<<<dim3(NTOK * NHc), dim3(128), 0, stream>>>(vraw, conv_v, vn, 0);

    // 4) low-rank decay gate: gf = (x Wfa^T) Wfb^T ; g = -exp(A_log)*softplus(gf+dt_bias)
    gemm(xh, wfah, nullptr, xah, NTOK, Dc, Hc);
    gemm(xah, wfbh, gf, nullptr, NTOK, Pc, Dc);
    {
        long n = (long)NTOK * Pc;
        gate_decay<<<dim3((unsigned)((n + 255) / 256)), dim3(256), 0, stream>>>(
            gf, A_log, dt_bias, gdec, n);
    }

    // 5) low-rank output gate: gate = (x Wga^T) Wgb^T
    gemm(xh, wgah, nullptr, xgh, NTOK, Dc, Hc);
    gemm(xgh, wgbh, gate, nullptr, NTOK, Pc, Dc);

    // 6) beta = sigmoid(x Wb^T)
    beta_kernel<<<dim3(NTOK * NHc / 8), dim3(256), 0, stream>>>(x, Wb, beta);

    // 7) sequential gated delta-rule recurrence
    kda_recurrence<<<dim3(Bc * NHc), dim3(128), 0, stream>>>(qn, kn, vn, gdec, beta, obuf);

    // 8) gated RMSNorm -> f16
    out_gate_norm<<<dim3(NTOK * NHc), dim3(128), 0, stream>>>(obuf, gate, norm_w, ogh);

    // 9) output projection (WMMA) -> d_out f32
    gemm(ogh, woh, out, nullptr, NTOK, Hc, Pc);
}
